// Attention_33543694582352
// MI455X (gfx1250) — compile-verified, matrix operand order checked by reading, not run on hardware
//
#include <hip/hip_runtime.h>
#include <hip/hip_bf16.h>

// ---------------------------------------------------------------------------
// Cross-attention B=2, S=2048, D=1024, H=16, HD=64 on gfx1250 (MI455X).
// fp32 -> bf16, all GEMMs via v_wmma_f32_16x16x32_bf16. Flash attention with
// cooperative K/V staging into LDS through the CDNA5 async-load path
// (global_load_async_to_lds_b128 + s_wait_asynccnt), double-buffered.
// ---------------------------------------------------------------------------

namespace {

constexpr int Sc  = 2048;
constexpr int Dc  = 1024;
constexpr int Hc  = 16;
constexpr int HDc = 64;
constexpr int LDK = 72;          // LDS row pitch in bf16 (64 data + 8 pad)

typedef __attribute__((ext_vector_type(16))) __bf16 bf16x16;
typedef __attribute__((ext_vector_type(8)))  float  f32x8;

union FragU {
    bf16x16 v;
    __bf16  h[16];
    uint4   q[2];
};

__device__ __forceinline__ f32x8 wmma_bf16(bf16x16 a, bf16x16 b, f32x8 c) {
    return __builtin_amdgcn_wmma_f32_16x16x32_bf16(
        false, a, false, b, (short)0, c, false, false);
}

__device__ __forceinline__ unsigned short f2bf(float f) {
    union { __bf16 h; unsigned short s; } u;
    u.h = (__bf16)f;
    return u.s;
}

// 16x32 A-frag / 32x16 B-frag from row-major bf16 (global or LDS).
// Lane L: row (L&15); k-chunks {(L>>4)*8..+7} and {16+(L>>4)*8..+7}.
__device__ __forceinline__ bf16x16 load_frag_bf16(const unsigned short* p,
                                                  int stride, int lane) {
    const unsigned short* rowp = p + (lane & 15) * stride;
    const int ks = (lane >> 4) << 3;
    FragU f;
    f.q[0] = *reinterpret_cast<const uint4*>(rowp + ks);
    f.q[1] = *reinterpret_cast<const uint4*>(rowp + 16 + ks);
    return f.v;
}

// Same fragment but sourced from fp32 with inline convert (projection only).
__device__ __forceinline__ bf16x16 load_frag_f32(const float* __restrict__ p,
                                                 int stride, int lane) {
    const float* rowp = p + (lane & 15) * stride;
    const int ks = (lane >> 4) << 3;
    float4 x0 = *reinterpret_cast<const float4*>(rowp + ks);
    float4 x1 = *reinterpret_cast<const float4*>(rowp + ks + 4);
    float4 x2 = *reinterpret_cast<const float4*>(rowp + 16 + ks);
    float4 x3 = *reinterpret_cast<const float4*>(rowp + 16 + ks + 4);
    FragU f;
    f.h[0]  = (__bf16)x0.x; f.h[1]  = (__bf16)x0.y;
    f.h[2]  = (__bf16)x0.z; f.h[3]  = (__bf16)x0.w;
    f.h[4]  = (__bf16)x1.x; f.h[5]  = (__bf16)x1.y;
    f.h[6]  = (__bf16)x1.z; f.h[7]  = (__bf16)x1.w;
    f.h[8]  = (__bf16)x2.x; f.h[9]  = (__bf16)x2.y;
    f.h[10] = (__bf16)x2.z; f.h[11] = (__bf16)x2.w;
    f.h[12] = (__bf16)x3.x; f.h[13] = (__bf16)x3.y;
    f.h[14] = (__bf16)x3.z; f.h[15] = (__bf16)x3.w;
    return f.v;
}

// CDNA5 async DMA: one 16-byte global -> LDS transfer per lane (ASYNCcnt).
__device__ __forceinline__ void async_b128(unsigned lds_addr,
                                           const unsigned short* g) {
    asm volatile("global_load_async_to_lds_b128 %0, %1, off"
                 :: "v"(lds_addr),
                    "v"((unsigned long long)(uintptr_t)g)
                 : "memory");
}

__device__ __forceinline__ void wait_async0() {
    asm volatile("s_wait_asynccnt 0x0" ::: "memory");
}

// ---------------------------------------------------------------------------
// Projection: Out = X(BS x D) @ W(D x D)^T, bf16 output, head-major:
//   VT=false : Out[((b*H+h)*S + s)*64 + hd]   (Q / K)
//   VT=true  : Out[((b*H+h)*64 + hd)*S + s]   (V transposed)
// ---------------------------------------------------------------------------
template <bool VT>
__global__ __launch_bounds__(256)
void proj_kernel(const float* __restrict__ X, const float* __restrict__ W,
                 unsigned short* __restrict__ Out, float scale) {
    const int lane = threadIdx.x & 31;
    const int wave = threadIdx.x >> 5;
    const int row0 = blockIdx.x * 256 + wave * 32;
    const int col0 = blockIdx.y * 64;

    f32x8 c[2][4] = {};

#pragma unroll 1
    for (int k = 0; k < Dc; k += 32) {
        bf16x16 a0 = load_frag_f32(X + (size_t)row0 * Dc + k, Dc, lane);
        bf16x16 a1 = load_frag_f32(X + (size_t)(row0 + 16) * Dc + k, Dc, lane);
        bf16x16 b0 = load_frag_f32(W + (size_t)(col0 +  0) * Dc + k, Dc, lane);
        bf16x16 b1 = load_frag_f32(W + (size_t)(col0 + 16) * Dc + k, Dc, lane);
        bf16x16 b2 = load_frag_f32(W + (size_t)(col0 + 32) * Dc + k, Dc, lane);
        bf16x16 b3 = load_frag_f32(W + (size_t)(col0 + 48) * Dc + k, Dc, lane);
        c[0][0] = wmma_bf16(a0, b0, c[0][0]);
        c[0][1] = wmma_bf16(a0, b1, c[0][1]);
        c[0][2] = wmma_bf16(a0, b2, c[0][2]);
        c[0][3] = wmma_bf16(a0, b3, c[0][3]);
        c[1][0] = wmma_bf16(a1, b0, c[1][0]);
        c[1][1] = wmma_bf16(a1, b1, c[1][1]);
        c[1][2] = wmma_bf16(a1, b2, c[1][2]);
        c[1][3] = wmma_bf16(a1, b3, c[1][3]);
    }

    const int rsel = (lane >> 4) << 3;
    const int nsel = lane & 15;
#pragma unroll
    for (int mi = 0; mi < 2; ++mi)
#pragma unroll
        for (int j = 0; j < 4; ++j)
#pragma unroll
            for (int r = 0; r < 8; ++r) {
                const int row = row0 + mi * 16 + r + rsel;
                const int o   = col0 + j * 16 + nsel;
                const int b   = row >> 11;
                const int s   = row & (Sc - 1);
                const int h   = o >> 6;
                const int hd  = o & (HDc - 1);
                size_t addr;
                if (VT)
                    addr = ((size_t)(b * Hc + h) * HDc + hd) * Sc + s;
                else
                    addr = ((size_t)(b * Hc + h) * Sc + s) * HDc + hd;
                Out[addr] = f2bf(c[mi][j][r] * scale);
            }
}

// ---------------------------------------------------------------------------
// Flash attention: 256 threads = 8 waves; each wave owns 16 query rows
// (128 rows / block) of one (b,h). Keys streamed in tiles of 64 through a
// double-buffered LDS stage filled by async DMA. Q pre-scaled by
// HD^-1/2 * log2(e) so softmax runs in the exp2 domain.
// ---------------------------------------------------------------------------
__global__ __launch_bounds__(256)
void attn_kernel(const unsigned short* __restrict__ Q,
                 const unsigned short* __restrict__ K,
                 const unsigned short* __restrict__ Vt,
                 float* __restrict__ out) {
    __shared__ __align__(16) unsigned short Ktile[2][64][LDK];  // [buf][key][hd]
    __shared__ __align__(16) unsigned short Vtile[2][64][LDK];  // [buf][hd][key]
    __shared__ __align__(16) unsigned short Ptile[8][16][LDK];  // per-wave P

    const int tid  = threadIdx.x;
    const int lane = tid & 31;
    const int wave = tid >> 5;
    const int bh   = blockIdx.y;
    const int b    = bh >> 4;
    const int h    = bh & (Hc - 1);
    const int q0   = blockIdx.x * 128 + wave * 16;

    const unsigned short* Qb  = Q  + ((size_t)bh * Sc + q0) * HDc;
    const unsigned short* Kb  = K  + (size_t)bh * Sc * HDc;
    const unsigned short* Vtb = Vt + (size_t)bh * HDc * Sc;

    // Per-thread DMA assignment: 2 x 16B chunks of K and of V per tile.
    // chunk c in [0,512): row = c>>3, 16B-column = c&7.
    const int c0 = tid * 2, c1 = tid * 2 + 1;
    const int kr0 = c0 >> 3, kc0 = (c0 & 7) * 8;
    const int kr1 = c1 >> 3, kc1 = (c1 & 7) * 8;

    auto issue_tile = [&](int kv, int buf) {
        async_b128((unsigned)(uintptr_t)&Ktile[buf][kr0][kc0],
                   Kb + (size_t)(kv + kr0) * HDc + kc0);
        async_b128((unsigned)(uintptr_t)&Ktile[buf][kr1][kc1],
                   Kb + (size_t)(kv + kr1) * HDc + kc1);
        async_b128((unsigned)(uintptr_t)&Vtile[buf][kr0][kc0],
                   Vtb + (size_t)kr0 * Sc + kv + kc0);
        async_b128((unsigned)(uintptr_t)&Vtile[buf][kr1][kc1],
                   Vtb + (size_t)kr1 * Sc + kv + kc1);
    };

    // Q tile 16x64 -> two resident A fragments.
    const bf16x16 a0 = load_frag_bf16(Qb,      HDc, lane);
    const bf16x16 a1 = load_frag_bf16(Qb + 32, HDc, lane);

    float m[8], l[8];
    f32x8 acc[4] = {};
#pragma unroll
    for (int r = 0; r < 8; ++r) { m[r] = -3.0e38f; l[r] = 0.0f; }

    const int rsel = (lane >> 4) << 3;
    const int nsel = lane & 15;

    issue_tile(0, 0);

#pragma unroll 1
    for (int t = 0; t < Sc / 64; ++t) {
        const int cur = t & 1;
        wait_async0();        // my DMA for buffer `cur` has landed
        __syncthreads();      // everyone's has; prev reads of other buf done
        if (t + 1 < Sc / 64) issue_tile((t + 1) * 64, cur ^ 1);

        // ---- scores: 16 queries x 64 keys (4 C frags), K-dim = 64 ----
        f32x8 s[4];
#pragma unroll
        for (int j = 0; j < 4; ++j) {
            bf16x16 k0 = load_frag_bf16(&Ktile[cur][16 * j][0],  LDK, lane);
            bf16x16 k1 = load_frag_bf16(&Ktile[cur][16 * j][32], LDK, lane);
            f32x8 z = {};
            s[j] = wmma_bf16(a0, k0, z);
            s[j] = wmma_bf16(a1, k1, s[j]);
        }

        // ---- online softmax in exp2 domain (row-wise, 16-lane halves) ----
        float tm[8];
#pragma unroll
        for (int r = 0; r < 8; ++r)
            tm[r] = fmaxf(fmaxf(s[0][r], s[1][r]), fmaxf(s[2][r], s[3][r]));
#pragma unroll
        for (int mask = 8; mask >= 1; mask >>= 1)
#pragma unroll
            for (int r = 0; r < 8; ++r)
                tm[r] = fmaxf(tm[r], __shfl_xor(tm[r], mask, 32));

        float alpha[8], rs[8];
#pragma unroll
        for (int r = 0; r < 8; ++r) {
            const float nm = fmaxf(m[r], tm[r]);
            alpha[r] = __builtin_amdgcn_exp2f(m[r] - nm);
            m[r] = nm;
            rs[r] = 0.0f;
#pragma unroll
            for (int j = 0; j < 4; ++j) {
                s[j][r] = __builtin_amdgcn_exp2f(s[j][r] - nm);
                rs[r] += s[j][r];
            }
        }
#pragma unroll
        for (int mask = 8; mask >= 1; mask >>= 1)
#pragma unroll
            for (int r = 0; r < 8; ++r)
                rs[r] += __shfl_xor(rs[r], mask, 32);
#pragma unroll
        for (int r = 0; r < 8; ++r) l[r] = l[r] * alpha[r] + rs[r];
#pragma unroll
        for (int j = 0; j < 4; ++j)
#pragma unroll
            for (int r = 0; r < 8; ++r) acc[j][r] *= alpha[r];

        // ---- P (f32 C frags) -> bf16 A frags via per-wave LDS stage ----
#pragma unroll
        for (int j = 0; j < 4; ++j)
#pragma unroll
            for (int r = 0; r < 8; ++r)
                Ptile[wave][r + rsel][16 * j + nsel] = f2bf(s[j][r]);
        __asm__ volatile("" ::: "memory");   // keep intra-wave store->load order
        const bf16x16 pa0 = load_frag_bf16(&Ptile[wave][0][0],  LDK, lane);
        const bf16x16 pa1 = load_frag_bf16(&Ptile[wave][0][32], LDK, lane);

        // ---- O += P(16x64) @ V(64x64) ----
#pragma unroll
        for (int j = 0; j < 4; ++j) {
            bf16x16 v0 = load_frag_bf16(&Vtile[cur][16 * j][0],  LDK, lane);
            bf16x16 v1 = load_frag_bf16(&Vtile[cur][16 * j][32], LDK, lane);
            acc[j] = wmma_bf16(pa0, v0, acc[j]);
            acc[j] = wmma_bf16(pa1, v1, acc[j]);
        }

        __syncthreads();      // all waves done with buffer `cur`
    }

    // ---- normalize and write (B, S, D) fp32 ----
#pragma unroll
    for (int j = 0; j < 4; ++j)
#pragma unroll
        for (int r = 0; r < 8; ++r) {
            const int row = q0 + r + rsel;
            const int col = h * HDc + j * 16 + nsel;
            out[((size_t)b * Sc + row) * Dc + col] = acc[j][r] / l[r];
        }
}

}  // namespace

// ---------------------------------------------------------------------------
// Host entry. Inputs: x[B,S,D] f32, context[B,S,D] f32, Wq[D,D] f32,
// Wkv[2D,D] f32. Output: [B,S,D] f32. Workspace: 24 MB of bf16 Q/K/Vt.
// ---------------------------------------------------------------------------
extern "C" void kernel_launch(void* const* d_in, const int* in_sizes, int n_in,
                              void* d_out, int out_size, void* d_ws, size_t ws_size,
                              hipStream_t stream) {
    (void)in_sizes; (void)n_in; (void)out_size; (void)ws_size;
    const float* x   = (const float*)d_in[0];
    const float* ctx = (const float*)d_in[1];
    const float* Wq  = (const float*)d_in[2];
    const float* Wkv = (const float*)d_in[3];
    float* out = (float*)d_out;

    const size_t elems = (size_t)2 * Sc * Dc;
    unsigned short* Qb  = (unsigned short*)d_ws;
    unsigned short* Kb  = Qb + elems;
    unsigned short* Vtb = Kb + elems;

    // Q scale folds HD^-1/2 and log2(e) (softmax runs in exp2 domain).
    const float qscale = 0.125f * 1.4426950408889634f;

    dim3 pgrid(16, 16);
    proj_kernel<false><<<pgrid, 256, 0, stream>>>(x,   Wq,            Qb,  qscale);
    proj_kernel<false><<<pgrid, 256, 0, stream>>>(ctx, Wkv,           Kb,  1.0f);
    proj_kernel<true ><<<pgrid, 256, 0, stream>>>(ctx, Wkv + Dc * Dc, Vtb, 1.0f);

    dim3 agrid(Sc / 128, 32);   // (16, B*H)
    attn_kernel<<<agrid, 256, 0, stream>>>(Qb, Kb, Vtb, out);
}
